// PMNet_16535624089838
// MI455X (gfx1250) — compile-verified, matrix operand order checked by reading, not run on hardware
//
#include <hip/hip_runtime.h>
#include <hip/hip_bf16.h>
#include <hip/hip_fp16.h>
#include <math.h>

// ---------------------------------------------------------------------------
// PMNet forward for MI455X (gfx1250, wave32, WMMA + async LDS copy).
// GEMM chain 32768x{96,512}x{512,96} on v_wmma_f32_16x16x32_f16.
// B operand staged per-workgroup through LDS via global_load_async_to_lds_b128
// (ASYNCcnt) so the 8 waves share one copy; A streams direct from global.
// ---------------------------------------------------------------------------

typedef __attribute__((ext_vector_type(16))) _Float16 v16h;
typedef __attribute__((ext_vector_type(8)))  float    v8f;

#define NJ     22
#define BS     64
#define TT     512
#define MROWS  (BS * TT)      // 32768
#define HMC    128

static __device__ __forceinline__ void wait_async0() {
#if defined(__has_builtin) && __has_builtin(__builtin_amdgcn_s_wait_asynccnt)
    __builtin_amdgcn_s_wait_asynccnt(0);
#else
    asm volatile("s_wait_asynccnt 0x0" ::: "memory");
#endif
}

// ---------------- weight pack: W[K][Nsrc] (f32) -> Wt[Npad][Kpad] (f16) -----
__global__ void pack_w_kernel(const float* __restrict__ W, _Float16* __restrict__ Wt,
                              int K, int N, int Nsrc, int Kpad, int Npad) {
    int idx = blockIdx.x * blockDim.x + threadIdx.x;
    if (idx >= Kpad * Npad) return;
    int n = idx / Kpad, k = idx % Kpad;
    float v = (n < N && k < K) ? W[(size_t)k * Nsrc + n] : 0.0f;
    Wt[(size_t)n * Kpad + k] = (_Float16)v;
}

// ---------------- x pack: seqA[..., :66] -> f16 [MROWS][96] (zero padded) ---
__global__ void pack_x_kernel(const float* __restrict__ seqA, _Float16* __restrict__ x16) {
    int idx = blockIdx.x * blockDim.x + threadIdx.x;
    if (idx >= MROWS * 96) return;
    int r = idx / 96, k = idx % 96;
    float v = (k < 66) ? seqA[(size_t)r * 70 + k] : 0.0f;
    x16[idx] = (_Float16)v;
}

// ---------------- ref_emb = sigmoid(skelB[:,0,:] @ ref_w + ref_b) -----------
__global__ void ref_emb_kernel(const float* __restrict__ skelB,
                               const float* __restrict__ ref_w,
                               const float* __restrict__ ref_b,
                               float* __restrict__ remb) {
    int idx = blockIdx.x * blockDim.x + threadIdx.x;
    if (idx >= BS * 16) return;
    int b = idx / 16, e = idx % 16;
    const float* s = skelB + (size_t)b * TT * 66;   // t = 0 row
    float acc = ref_b[e];
    for (int c = 0; c < 66; ++c) acc += s[c] * ref_w[c * 16 + e];
    remb[idx] = 1.0f / (1.0f + __expf(-acc));
}

// ---------------- dbias[b][n] = d_b[n] + ref_emb[b] @ d_w[512:528] ----------
__global__ void dbias_kernel(const float* __restrict__ remb,
                             const float* __restrict__ d_w,
                             const float* __restrict__ d_b,
                             float* __restrict__ dbias) {
    int idx = blockIdx.x * blockDim.x + threadIdx.x;
    if (idx >= BS * 88) return;
    int b = idx / 88, n = idx % 88;
    float acc = d_b[n];
    #pragma unroll
    for (int e = 0; e < 16; ++e) acc += remb[b * 16 + e] * d_w[(512 + e) * 88 + n];
    dbias[idx] = acc;
}

// ---------------- WMMA GEMM with async-LDS-staged B -------------------------
// Block = 256 threads = 8 waves; block tile 256 rows x NT*16 cols.
// Wave tile: 32 x NT*16 (2 A frags, NT B frags, 2*NT WMMAs per K-step).
// Per 32-wide K chunk, threads cooperatively issue
// global_load_async_to_lds_b128 for the shared B tile (NT*1024 bytes).
template <int NT>
__global__ __launch_bounds__(256)
void gemm_wmma_lds(const _Float16* __restrict__ A, int lda,
                   const _Float16* __restrict__ Bt, int ldb, int K,
                   const float* __restrict__ bias, int biasMode,
                   int act,
                   _Float16* __restrict__ out16, int ldo16,
                   float* __restrict__ out32, int ldo32, int nvalid) {
    __shared__ __align__(16) _Float16 sB[NT * 16 * 32];   // [col][32 K halves]

    int tid  = threadIdx.x;
    int wave = tid >> 5;
    int lane = tid & 31;
    int m    = lane & 15;
    int hi   = lane >> 4;
    int row0 = blockIdx.y * 256 + wave * 32;
    int colB = blockIdx.x * (NT * 16);

    const _Float16* ap0 = A + (size_t)(row0 + m) * lda + hi * 8;
    const _Float16* ap1 = A + (size_t)(row0 + 16 + m) * lda + hi * 8;

    // async-copy lane assignment: 4 threads per column, 16B each
    const char* gB  = (const char*)(Bt + (size_t)colB * ldb);
    int  cc = tid >> 2;
    int  oo = (tid & 3) * 16;
    unsigned lbase = (unsigned)(unsigned long long)(void*)&sB[0];
    unsigned laddr = lbase + (unsigned)(cc * 64 + oo);

    v8f acc0[NT], acc1[NT];
    #pragma unroll
    for (int nt = 0; nt < NT; ++nt) { acc0[nt] = (v8f){}; acc1[nt] = (v8f){}; }

    union Frag { v16h v; uint4 q[2]; };

    for (int kc = 0; kc < K; kc += 32) {
        __syncthreads();                       // previous chunk fully consumed
        if (tid < NT * 64) {
            unsigned long long g =
                (unsigned long long)(gB + (size_t)cc * ldb * 2 + (size_t)kc * 2 + oo);
            asm volatile("global_load_async_to_lds_b128 %0, %1, off"
                         :: "v"(laddr), "v"(g) : "memory");
        }
        wait_async0();
        __syncthreads();                       // B chunk visible to all waves

        Frag fa0, fa1;
        fa0.q[0] = *(const uint4*)(ap0 + kc);
        fa0.q[1] = *(const uint4*)(ap0 + kc + 16);
        fa1.q[0] = *(const uint4*)(ap1 + kc);
        fa1.q[1] = *(const uint4*)(ap1 + kc + 16);

        Frag fb[NT];
        #pragma unroll
        for (int nt = 0; nt < NT; ++nt) {
            const _Float16* bp = sB + (nt * 16 + m) * 32 + hi * 8;
            fb[nt].q[0] = *(const uint4*)(bp);
            fb[nt].q[1] = *(const uint4*)(bp + 16);
        }

        #pragma unroll
        for (int nt = 0; nt < NT; ++nt) {
            acc0[nt] = __builtin_amdgcn_wmma_f32_16x16x32_f16(false, fa0.v, false, fb[nt].v,
                                                              (short)0, acc0[nt], false, false);
            acc1[nt] = __builtin_amdgcn_wmma_f32_16x16x32_f16(false, fa1.v, false, fb[nt].v,
                                                              (short)0, acc1[nt], false, false);
        }
    }

    #pragma unroll
    for (int mt = 0; mt < 2; ++mt) {
        #pragma unroll
        for (int nt = 0; nt < NT; ++nt) {
            v8f acc = mt ? acc1[nt] : acc0[nt];
            int col = colB + nt * 16 + m;
            bool colOK = col < nvalid;
            #pragma unroll
            for (int e = 0; e < 8; ++e) {
                int row = row0 + mt * 16 + hi * 8 + e;
                float v = acc[e];
                if (bias && colOK)
                    v += (biasMode == 0) ? bias[col] : bias[(row >> 9) * 88 + col];
                if (act) v = v > 0.0f ? v : 0.0f;
                if (out16 && colOK) out16[(size_t)row * ldo16 + col] = (_Float16)v;
                if (out32 && colOK) out32[(size_t)row * ldo32 + col] = v;
            }
        }
    }
}

// ---------------- quaternion normalize, qB = qA*dq, FK chain ----------------
__device__ __forceinline__ void qmul(const float a[4], const float b[4], float o[4]) {
    o[0] = a[0]*b[0] - a[1]*b[1] - a[2]*b[2] - a[3]*b[3];
    o[1] = a[0]*b[1] + a[1]*b[0] + a[2]*b[3] - a[3]*b[2];
    o[2] = a[0]*b[2] - a[1]*b[3] + a[2]*b[0] + a[3]*b[1];
    o[3] = a[0]*b[3] + a[1]*b[2] - a[2]*b[1] + a[3]*b[0];
}
__device__ __forceinline__ void qrot(const float q[4], const float v[3], float o[3]) {
    float t0 = 2.0f * (q[2]*v[2] - q[3]*v[1]);
    float t1 = 2.0f * (q[3]*v[0] - q[1]*v[2]);
    float t2 = 2.0f * (q[1]*v[1] - q[2]*v[0]);
    o[0] = v[0] + q[0]*t0 + (q[2]*t2 - q[3]*t1);
    o[1] = v[1] + q[0]*t1 + (q[3]*t0 - q[1]*t2);
    o[2] = v[2] + q[0]*t2 + (q[1]*t1 - q[2]*t0);
}

__global__ void quat_fk_kernel(const float* __restrict__ qraw,
                               const float* __restrict__ draw,
                               const float* __restrict__ skelB,
                               const float* __restrict__ lmean,
                               const float* __restrict__ lstd,
                               float* __restrict__ outLocal,
                               float* __restrict__ outQB,
                               float* __restrict__ outDQ) {
    int idx = blockIdx.x * blockDim.x + threadIdx.x;
    if (idx >= MROWS) return;
    int b = idx >> 9;
    constexpr int PAR[NJ] = {-1,0,1,2,3,0,5,6,7,0,9,10,11,10,13,14,15,10,17,18,19,10};

    const float* qr = qraw + (size_t)idx * 88;
    const float* dr = draw + (size_t)idx * 88;
    float qB[NJ][4];
    #pragma unroll
    for (int j = 0; j < NJ; ++j) {
        float qa[4], dq[4];
        float sq = 0.f, sd = 0.f;
        #pragma unroll
        for (int c = 0; c < 4; ++c) { qa[c] = qr[j*4+c]; sq += qa[c]*qa[c];
                                      dq[c] = dr[j*4+c]; sd += dq[c]*dq[c]; }
        float rq = rsqrtf(sq), rd = rsqrtf(sd);
        #pragma unroll
        for (int c = 0; c < 4; ++c) { qa[c] *= rq; dq[c] *= rd; }
        #pragma unroll
        for (int c = 0; c < 4; ++c) outDQ[((size_t)idx * NJ + j) * 4 + c] = dq[c];
        qmul(qa, dq, qB[j]);
        #pragma unroll
        for (int c = 0; c < 4; ++c) outQB[((size_t)idx * NJ + j) * 4 + c] = qB[j][c];
    }

    float refB[NJ][3];
    const float* s = skelB + (size_t)b * TT * 66;   // t = 0 row
    #pragma unroll
    for (int j = 0; j < NJ; ++j)
        #pragma unroll
        for (int c = 0; c < 3; ++c)
            refB[j][c] = s[j*3+c] * lstd[j*3+c] + lmean[j*3+c];

    float gq[NJ][4], gp[NJ][3];
    #pragma unroll
    for (int c = 0; c < 4; ++c) gq[0][c] = qB[0][c];
    #pragma unroll
    for (int c = 0; c < 3; ++c) gp[0][c] = refB[0][c];
    #pragma unroll
    for (int j = 1; j < NJ; ++j) {
        int p = PAR[j];
        float off[3] = { refB[j][0]-refB[p][0], refB[j][1]-refB[p][1], refB[j][2]-refB[p][2] };
        qmul(gq[p], qB[j], gq[j]);
        float r[3];
        qrot(gq[p], off, r);
        gp[j][0] = gp[p][0] + r[0]; gp[j][1] = gp[p][1] + r[1]; gp[j][2] = gp[p][2] + r[2];
    }
    #pragma unroll
    for (int j = 0; j < NJ; ++j)
        #pragma unroll
        for (int c = 0; c < 3; ++c)
            outLocal[((size_t)idx * NJ + j) * 3 + c] = (gp[j][c] - lmean[j*3+c]) / lstd[j*3+c];
}

// ---------------- conv1: vin(4ch) -> 128ch, taps 3, leaky 0.2 ---------------
__global__ void conv1_kernel(const float* __restrict__ seqA,
                             const float* __restrict__ inp_h,
                             const float* __restrict__ w1,
                             const float* __restrict__ b1,
                             float* __restrict__ m1) {
    int idx = blockIdx.x * blockDim.x + threadIdx.x;
    if (idx >= BS * HMC * TT) return;
    int t  = idx % TT;
    int hm = (idx / TT) % HMC;
    int b  = idx / (TT * HMC);
    float ihr = 1.0f / inp_h[b];
    float acc = b1[hm];
    #pragma unroll
    for (int k = 0; k < 3; ++k) {
        int tt = t + k - 1;
        if (tt < 0 || tt >= TT) continue;
        const float* row = seqA + ((size_t)b * TT + tt) * 70 + 66;
        const float* w = w1 + hm * 12 + k;          // w1[hm][c][k]
        acc += w[0] * (row[0] * ihr) + w[3] * (row[1] * ihr)
             + w[6] * (row[2] * ihr) + w[9] *  row[3];
    }
    m1[idx] = acc > 0.0f ? acc : 0.2f * acc;        // layout [b][hm][t]
}

// ---------------- conv2: 128ch -> 4ch, taps 3, height scale -----------------
__global__ void conv2_kernel(const float* __restrict__ m1,
                             const float* __restrict__ w2,
                             const float* __restrict__ b2,
                             const float* __restrict__ tgt_h,
                             float* __restrict__ outG) {
    int idx = blockIdx.x * blockDim.x + threadIdx.x;
    if (idx >= BS * TT * 4) return;
    int co = idx & 3;
    int t  = (idx >> 2) % TT;
    int b  = idx >> 11;
    float acc = b2[co];
    const float* mb = m1 + (size_t)b * HMC * TT;
    for (int hm = 0; hm < HMC; ++hm) {
        const float* w  = w2 + (co * HMC + hm) * 3; // w2[co][hm][k]
        const float* mm = mb + hm * TT + t;
        if (t >= 1)      acc += w[0] * mm[-1];
        acc += w[1] * mm[0];
        if (t < TT - 1)  acc += w[2] * mm[1];
    }
    if (co < 3) acc *= tgt_h[b];
    outG[idx] = acc;                                // (b,t,4) layout
}

// ---------------------------------------------------------------------------
extern "C" void kernel_launch(void* const* d_in, const int* in_sizes, int n_in,
                              void* d_out, int out_size, void* d_ws, size_t ws_size,
                              hipStream_t stream) {
    (void)in_sizes; (void)n_in; (void)out_size; (void)ws_size;
    const float* seqA   = (const float*)d_in[0];
    const float* skelB  = (const float*)d_in[3];
    const float* inp_h  = (const float*)d_in[4];
    const float* tgt_h  = (const float*)d_in[5];
    const float* lmean  = (const float*)d_in[6];
    const float* lstd   = (const float*)d_in[7];
    const float* p_w1   = (const float*)d_in[9];
    const float* p_b1   = (const float*)d_in[10];
    const float* p_w2   = (const float*)d_in[11];
    const float* p_b2   = (const float*)d_in[12];
    const float* p_w3   = (const float*)d_in[13];
    const float* p_b3   = (const float*)d_in[14];
    const float* p_w4   = (const float*)d_in[15];
    const float* p_b4   = (const float*)d_in[16];
    const float* q_w    = (const float*)d_in[17];
    const float* q_b    = (const float*)d_in[18];
    const float* ref_w  = (const float*)d_in[19];
    const float* ref_b  = (const float*)d_in[20];
    const float* d_w    = (const float*)d_in[21];
    const float* d_b    = (const float*)d_in[22];
    const float* m_w1   = (const float*)d_in[23];
    const float* m_b1   = (const float*)d_in[24];
    const float* m_w2   = (const float*)d_in[25];
    const float* m_b2   = (const float*)d_in[26];

    char* ws = (char*)d_ws;
    _Float16* x16  = (_Float16*)(ws + 0);            // 32768*96  f16
    _Float16* hA   = (_Float16*)(ws + 6291456);      // 32768*512 f16
    _Float16* hB   = (_Float16*)(ws + 39845888);     // 32768*512 f16
    _Float16* w1t  = (_Float16*)(ws + 73400320);     // 512*96
    _Float16* w2t  = (_Float16*)(ws + 73498624);     // 512*512
    _Float16* w3t  = (_Float16*)(ws + 74022912);
    _Float16* w4t  = (_Float16*)(ws + 74547200);
    _Float16* qwt  = (_Float16*)(ws + 75071488);     // 96*512
    _Float16* dwt  = (_Float16*)(ws + 75169792);     // 96*512
    float*    remb = (float*)   (ws + 75268096);     // 64*16
    float*    dbias= (float*)   (ws + 75272192);     // 64*88
    float*    qraw = (float*)   (ws + 75294720);     // 32768*88 f32
    float*    draw = (float*)   (ws + 86829056);     // 32768*88 f32
    float*    m1   = (float*)   (ws + 98363392);     // 64*128*512 f32

    float* out       = (float*)d_out;
    float* outLocal  = out;                 // 64*512*22*3
    float* outGlobal = out + 2162688;       // 64*512*4
    float* outQB     = out + 2293760;       // 64*512*22*4
    float* outDQ     = out + 5177344;       // 64*512*22*4

    const int B = 256;
    // weight / activation packing
    pack_w_kernel<<<(512*96  + B-1)/B, B, 0, stream>>>(p_w1, w1t,  66, 512, 512,  96, 512);
    pack_w_kernel<<<(512*512 + B-1)/B, B, 0, stream>>>(p_w2, w2t, 512, 512, 512, 512, 512);
    pack_w_kernel<<<(512*512 + B-1)/B, B, 0, stream>>>(p_w3, w3t, 512, 512, 512, 512, 512);
    pack_w_kernel<<<(512*512 + B-1)/B, B, 0, stream>>>(p_w4, w4t, 512, 512, 512, 512, 512);
    pack_w_kernel<<<(96*512  + B-1)/B, B, 0, stream>>>(q_w,  qwt, 512,  88,  88, 512,  96);
    pack_w_kernel<<<(96*512  + B-1)/B, B, 0, stream>>>(d_w,  dwt, 512,  88,  88, 512,  96);
    pack_x_kernel<<<(MROWS*96 + B-1)/B, B, 0, stream>>>(seqA, x16);
    ref_emb_kernel<<<(BS*16 + B-1)/B, B, 0, stream>>>(skelB, ref_w, ref_b, remb);
    dbias_kernel<<<(BS*88 + B-1)/B, B, 0, stream>>>(remb, d_w, d_b, dbias);

    // MLP chain on WMMA: block tile 256 x 64 (NT=4), grid = (N/64, M/256)
    gemm_wmma_lds<4><<<dim3( 8, 128), B, 0, stream>>>(x16,  96, w1t,  96,  96, p_b1, 0, 1, hA, 512, nullptr, 0, 512);
    gemm_wmma_lds<4><<<dim3( 8, 128), B, 0, stream>>>(hA,  512, w2t, 512, 512, p_b2, 0, 1, hB, 512, nullptr, 0, 512);
    gemm_wmma_lds<4><<<dim3( 8, 128), B, 0, stream>>>(hB,  512, w3t, 512, 512, p_b3, 0, 1, hA, 512, nullptr, 0, 512);
    gemm_wmma_lds<4><<<dim3( 8, 128), B, 0, stream>>>(hA,  512, w4t, 512, 512, p_b4, 0, 1, hB, 512, nullptr, 0, 512);
    // heads (fai = hB): block tile 256 x 32 (NT=2), grid = (96/32, 128)
    gemm_wmma_lds<2><<<dim3( 3, 128), B, 0, stream>>>(hB, 512, qwt, 512, 512, q_b,   0, 0, nullptr, 0, qraw, 88, 88);
    gemm_wmma_lds<2><<<dim3( 3, 128), B, 0, stream>>>(hB, 512, dwt, 512, 512, dbias, 1, 0, nullptr, 0, draw, 88, 88);

    // quaternions + FK
    quat_fk_kernel<<<(MROWS + B-1)/B, B, 0, stream>>>(qraw, draw, skelB, lmean, lstd,
                                                      outLocal, outQB, outDQ);
    // global trajectory convs
    conv1_kernel<<<(BS*HMC*TT + B-1)/B, B, 0, stream>>>(seqA, inp_h, m_w1, m_b1, m1);
    conv2_kernel<<<(BS*TT*4  + B-1)/B, B, 0, stream>>>(m1, m_w2, m_b2, tgt_h, outGlobal);
}
